// MultiHeadAttention_901943132503
// MI455X (gfx1250) — matrix operand order repaired and checked by reading, not compiled, not run on hardware
//
#include <hip/hip_runtime.h>
#include <hip/hip_bf16.h>

// ---------------------------------------------------------------------------
// MHA forward for MI455X (gfx1250, wave32, WMMA bf16 16x16x32, f32 accum)
//   0) cvt_f32_bf16 : one-time fp32 -> bf16 for activations + weights
//   1) gemm_bf16<MODE>: Y = X @ W^T + b, 32x64 wave tile, pure-bf16 operands
//        MODE 0: bf16 head-split [b,h,s,d] (Q scaled, K)
//        MODE 1: bf16 head-split transposed [b,h,d,s] (V^T)
//        MODE 2: fp32 row-major [b,s,D] (final projection -> d_out)
//   2) flash_attn: causal online-softmax attention, one wave per 16 q-rows
// ---------------------------------------------------------------------------

typedef __attribute__((ext_vector_type(16))) __bf16 v16bf;
typedef __attribute__((ext_vector_type(8)))  __bf16 bf16x8;
typedef __attribute__((ext_vector_type(8)))  float  v8f;
typedef __attribute__((ext_vector_type(4)))  float  f32x4;

constexpr int CB  = 2;       // batch
constexpr int CH  = 16;      // heads
constexpr int CS  = 2048;    // seq
constexpr int CHD = 128;     // head dim
constexpr int CD  = 2048;    // model dim
constexpr int CM  = CB * CS; // 4096 GEMM rows

__device__ __forceinline__ v8f wmma_bf16(v16bf a, v16bf b, v8f c) {
  return __builtin_amdgcn_wmma_f32_16x16x32_bf16(false, a, false, b,
                                                 (short)0, c, false, false);
}

// A-operand (16x32 bf16): lane row = lane&15; 16 contiguous k at hs*16.
// Caller passes p = row_base + hs*16.
__device__ __forceinline__ v16bf a_op_bf16(const __bf16* p) {
  bf16x8 lo = *(const bf16x8*)(p);
  bf16x8 hi = *(const bf16x8*)(p + 8);
  return __builtin_shufflevector(lo, hi, 0,1,2,3,4,5,6,7,8,9,10,11,12,13,14,15);
}
// B-operand (32x16 bf16): lane col = lane&15; 16 contiguous k at hs*16.
// Caller passes p = col_base + hs*16.
__device__ __forceinline__ v16bf b_op_bf16(const __bf16* p) {
  bf16x8 lo = *(const bf16x8*)(p);
  bf16x8 hi = *(const bf16x8*)(p + 8);
  return __builtin_shufflevector(lo, hi, 0,1,2,3,4,5,6,7,8,9,10,11,12,13,14,15);
}

// -------------------------- conversion pass --------------------------------
__global__ __launch_bounds__(256) void cvt_f32_bf16(const float* __restrict__ in,
                                                    __bf16* __restrict__ out,
                                                    int n) {
  const int i = (blockIdx.x * 256 + threadIdx.x) * 8; // n is a multiple of 2048
  if (i < n) {
    f32x4 x0 = *(const f32x4*)(in + i);
    f32x4 x1 = *(const f32x4*)(in + i + 4);
    bf16x8 o;
#pragma unroll
    for (int j = 0; j < 4; ++j) { o[j] = (__bf16)x0[j]; o[4 + j] = (__bf16)x1[j]; }
    *(bf16x8*)(out + i) = o;
  }
}

// ------------------------------- GEMM --------------------------------------
template <int MODE>
__global__ __launch_bounds__(256) void gemm_bf16(const __bf16* __restrict__ A,
                                                 const __bf16* __restrict__ W,
                                                 const float*  __restrict__ bias,
                                                 void* __restrict__ outp,
                                                 float scale) {
  const int lane = threadIdx.x & 31;
  const int lt   = lane & 15;
  const int hs   = lane >> 4;
  const int wid  = blockIdx.x * 8 + (threadIdx.x >> 5);
  constexpr int NT = CD / 64; // 32 n-tiles of 64
  const int m0 = (wid / NT) * 32;
  const int n0 = (wid % NT) * 64;

  v8f acc[2][4];
#pragma unroll
  for (int mi = 0; mi < 2; ++mi)
#pragma unroll
    for (int t = 0; t < 4; ++t)
#pragma unroll
      for (int r = 0; r < 8; ++r) acc[mi][t][r] = 0.0f;

  const __bf16* a0p = A + (size_t)(m0 + lt) * CD + hs * 16;
  const __bf16* a1p = A + (size_t)(m0 + 16 + lt) * CD + hs * 16;
  const __bf16* wp  = W + (size_t)(n0 + lt) * CD + hs * 16;

  for (int k0 = 0; k0 < CD; k0 += 32) {
    __builtin_prefetch(a0p + k0 + 256, 0, 0); // next-tile hint -> global_prefetch_b8
    v16bf a0 = a_op_bf16(a0p + k0);
    v16bf a1 = a_op_bf16(a1p + k0);
#pragma unroll
    for (int t = 0; t < 4; ++t) {
      // B[k][n] = W[n][k]: column n of B is row n of W, contiguous in k.
      v16bf b = b_op_bf16(wp + (size_t)t * 16 * CD + k0);
      acc[0][t] = wmma_bf16(a0, b, acc[0][t]);
      acc[1][t] = wmma_bf16(a1, b, acc[1][t]);
    }
  }

#pragma unroll
  for (int t = 0; t < 4; ++t) {
    const int   col = n0 + t * 16 + lt;
    const float bv  = bias[col];
#pragma unroll
    for (int mi = 0; mi < 2; ++mi)
#pragma unroll
      for (int r = 0; r < 8; ++r) {
        const int row = m0 + mi * 16 + r + hs * 8; // C/D: VGPR r -> rows r/r+8
        float v = (acc[mi][t][r] + bv) * scale;
        if (MODE == 2) {
          ((float*)outp)[(size_t)row * CD + col] = v;
        } else {
          const int bb = row / CS, ss = row % CS;
          const int hh = col / CHD, dd = col % CHD;
          __bf16* O = (__bf16*)outp;
          if (MODE == 0)
            O[(((size_t)bb * CH + hh) * CS + ss) * CHD + dd] = (__bf16)v;
          else
            O[(((size_t)bb * CH + hh) * CHD + dd) * CS + ss] = (__bf16)v;
        }
      }
  }
}

// --------------------------- flash attention -------------------------------
// Qh/Kh: [b,h,s,d] bf16 (Q pre-scaled by 1/sqrt(d)); Vt: [b,h,d,s] bf16.
// Oc: combined heads [b,s,D] bf16.
__global__ __launch_bounds__(256) void flash_attn(const __bf16* __restrict__ Qh,
                                                  const __bf16* __restrict__ Kh,
                                                  const __bf16* __restrict__ Vt,
                                                  __bf16* __restrict__ Oc) {
  __shared__ __bf16 pbuf[8][16 * 32]; // per-wave P-tile transpose staging

  const int wave = threadIdx.x >> 5;
  const int lane = threadIdx.x & 31;
  const int lt   = lane & 15;
  const int hs   = lane >> 4;
  const int wid  = blockIdx.x * 8 + wave;
  constexpr int QT = CS / 16;
  const int bh = wid / QT;
  const int q0 = (wid % QT) * 16;

  const __bf16* Qb = Qh + (size_t)bh * CS * CHD;
  const __bf16* Kb = Kh + (size_t)bh * CS * CHD;
  const __bf16* Vb = Vt + (size_t)bh * CHD * CS;

  // Q strip 16x128 as four A-operands (d chunks of 32)
  v16bf qa[4];
#pragma unroll
  for (int dc = 0; dc < 4; ++dc)
    qa[dc] = a_op_bf16(Qb + (size_t)(q0 + lt) * CHD + dc * 32 + hs * 16);

  v8f acc[8];
  float mrow[8], lrow[8];
#pragma unroll
  for (int dt = 0; dt < 8; ++dt)
#pragma unroll
    for (int r = 0; r < 8; ++r) acc[dt][r] = 0.0f;
#pragma unroll
  for (int r = 0; r < 8; ++r) { mrow[r] = -__builtin_inff(); lrow[r] = 0.0f; }

  __bf16* pb = &pbuf[wave][0];

  for (int j0 = 0; j0 < q0 + 16; j0 += 32) {
    const bool edge = (j0 + 31 > q0); // wave-uniform: only the diagonal block masks

    // S = Q K^T for 32 keys -> two 16x16 f32 tiles
    v8f s0, s1;
#pragma unroll
    for (int r = 0; r < 8; ++r) { s0[r] = 0.0f; s1[r] = 0.0f; }
#pragma unroll
    for (int dc = 0; dc < 4; ++dc) {
      v16bf kb0 = b_op_bf16(Kb + (size_t)(j0 + lt)      * CHD + dc * 32 + hs * 16);
      v16bf kb1 = b_op_bf16(Kb + (size_t)(j0 + 16 + lt) * CHD + dc * 32 + hs * 16);
      s0 = wmma_bf16(qa[dc], kb0, s0);
      s1 = wmma_bf16(qa[dc], kb1, s1);
    }

    float corr[8];
#pragma unroll
    for (int r = 0; r < 8; ++r) {
      const int row = q0 + r + hs * 8;
      float a0 = s0[r], a1 = s1[r];
      if (edge) {
        if (j0 + lt      > row) a0 = -__builtin_inff();
        if (j0 + 16 + lt > row) a1 = -__builtin_inff();
      }
      float mx = fmaxf(a0, a1);
      mx = fmaxf(mx, __shfl_xor(mx, 1, 32));
      mx = fmaxf(mx, __shfl_xor(mx, 2, 32));
      mx = fmaxf(mx, __shfl_xor(mx, 4, 32));
      mx = fmaxf(mx, __shfl_xor(mx, 8, 32));
      const float mn = fmaxf(mrow[r], mx);
      const float c  = __expf(mrow[r] - mn);
      mrow[r] = mn;
      const float p0 = __expf(a0 - mn);
      const float p1 = __expf(a1 - mn);
      // stage P in q-row-major bf16 for A-operand reload
      pb[(r + hs * 8) * 32 + lt]      = (__bf16)p0;
      pb[(r + hs * 8) * 32 + 16 + lt] = (__bf16)p1;
      float rs = p0 + p1;
      rs += __shfl_xor(rs, 1, 32);
      rs += __shfl_xor(rs, 2, 32);
      rs += __shfl_xor(rs, 4, 32);
      rs += __shfl_xor(rs, 8, 32);
      lrow[r] = lrow[r] * c + rs;
      corr[r] = c;
    }
#pragma unroll
    for (int dt = 0; dt < 8; ++dt)
#pragma unroll
      for (int r = 0; r < 8; ++r) acc[dt][r] *= corr[r];

    // P (16x32) as A-operand from LDS (wave-private; compiler inserts dscnt waits)
    v16bf pa = a_op_bf16(pb + lt * 32 + hs * 16);

    // acc += P @ V : V column d is row (d) of Vt, contiguous along keys
#pragma unroll
    for (int dt = 0; dt < 8; ++dt) {
      v16bf vb = b_op_bf16(Vb + (size_t)(dt * 16 + lt) * CS + j0 + hs * 16);
      acc[dt] = wmma_bf16(pa, vb, acc[dt]);
    }
  }

  const int bb = bh / CH, hh = bh % CH;
#pragma unroll
  for (int r = 0; r < 8; ++r) {
    const int   row = q0 + r + hs * 8;
    const float inv = 1.0f / lrow[r]; // one rcp per row, then multiplies
#pragma unroll
    for (int dt = 0; dt < 8; ++dt) {
      const int d = dt * 16 + lt;
      Oc[((size_t)(bb * CS + row)) * CD + hh * CHD + d] = (__bf16)(acc[dt][r] * inv);
    }
  }
}

// ------------------------------ launcher -----------------------------------
extern "C" void kernel_launch(void* const* d_in, const int* in_sizes, int n_in,
                              void* d_out, int out_size, void* d_ws, size_t ws_size,
                              hipStream_t stream) {
  (void)in_sizes; (void)n_in; (void)out_size; (void)ws_size;
  const float* q  = (const float*)d_in[0];
  const float* k  = (const float*)d_in[1];
  const float* v  = (const float*)d_in[2];
  // d_in[3] = tril mask (causality implemented analytically)
  const float* Wq = (const float*)d_in[4];
  const float* bq = (const float*)d_in[5];
  const float* Wk = (const float*)d_in[6];
  const float* bk = (const float*)d_in[7];
  const float* Wv = (const float*)d_in[8];
  const float* bv = (const float*)d_in[9];
  const float* Wo = (const float*)d_in[10];
  const float* bo = (const float*)d_in[11];

  const size_t nact = (size_t)CB * CS * CD; // 8,388,608
  const size_t nw   = (size_t)CD * CD;      // 4,194,304

  __bf16* p    = (__bf16*)d_ws;
  __bf16* xq   = p; p += nact;   // bf16 activations
  __bf16* xk   = p; p += nact;
  __bf16* xv   = p; p += nact;
  __bf16* wq   = p; p += nw;     // bf16 weights
  __bf16* wk   = p; p += nw;
  __bf16* wv   = p; p += nw;
  __bf16* wo   = p; p += nw;
  __bf16* qh   = p; p += nact;   // [b,h,s,d] (pre-scaled)
  __bf16* kh   = p; p += nact;   // [b,h,s,d]
  __bf16* vt   = p; p += nact;   // [b,h,d,s]
  __bf16* oc   = p; p += nact;   // [b,s,D]

  const int cvtA = (int)(nact / (8 * 256)); // 4096 blocks
  const int cvtW = (int)(nw   / (8 * 256)); // 2048 blocks
  cvt_f32_bf16<<<cvtA, 256, 0, stream>>>(q,  xq, (int)nact);
  cvt_f32_bf16<<<cvtA, 256, 0, stream>>>(k,  xk, (int)nact);
  cvt_f32_bf16<<<cvtA, 256, 0, stream>>>(v,  xv, (int)nact);
  cvt_f32_bf16<<<cvtW, 256, 0, stream>>>(Wq, wq, (int)nw);
  cvt_f32_bf16<<<cvtW, 256, 0, stream>>>(Wk, wk, (int)nw);
  cvt_f32_bf16<<<cvtW, 256, 0, stream>>>(Wv, wv, (int)nw);
  cvt_f32_bf16<<<cvtW, 256, 0, stream>>>(Wo, wo, (int)nw);

  const int gemm_blocks = (CM / 32) * (CD / 64) / 8; // 512 blocks x 256 thr
  const float scale = 0.08838834764831845f;          // 1/sqrt(128)

  gemm_bf16<0><<<gemm_blocks, 256, 0, stream>>>(xq, wq, bq, qh, scale);
  gemm_bf16<0><<<gemm_blocks, 256, 0, stream>>>(xk, wk, bk, kh, 1.0f);
  gemm_bf16<1><<<gemm_blocks, 256, 0, stream>>>(xv, wv, bv, vt, 1.0f);

  const int attn_blocks = CB * CH * (CS / 16) / 8;   // 512 blocks x 256 thr
  flash_attn<<<attn_blocks, 256, 0, stream>>>(qh, kh, vt, oc);

  gemm_bf16<2><<<gemm_blocks, 256, 0, stream>>>(oc, wo, bo, d_out, 1.0f);
}
